// GCN_LSTM_15779709846042
// MI455X (gfx1250) — matrix operand order repaired and checked by reading, not compiled, not run on hardware
//
#include <hip/hip_runtime.h>
#include <math.h>

// Problem sizes (fixed by the reference)
#define N_NODES 20000
#define SEQ_L   50
#define IN_DIM  128
#define H_DIM   64
#define G4H     256   // 4*H
#define NH_DIM  256
#define NC_DIM  16
#define VOCAB   32000
#define N_EDGES 640000

typedef __attribute__((ext_vector_type(16))) __bf16 v16bf;
typedef __attribute__((ext_vector_type(2)))  __bf16 v2bf;
typedef __attribute__((ext_vector_type(8)))  float  v8f;

union V16BF { v16bf v; unsigned int u[8]; };
union BF2   { v2bf v; unsigned int u; };

// Let the compiler pick the native cvt (v_cvt_pk_bf16_f32-class) for fptrunc.
__device__ __forceinline__ unsigned int pack_bf2(float lo, float hi) {
  BF2 r;
  r.v[0] = (__bf16)lo;
  r.v[1] = (__bf16)hi;
  return r.u;
}
__device__ __forceinline__ float sigf(float x) { return 1.0f / (1.0f + __expf(-x)); }

__device__ __forceinline__ v8f wmma_bf16(v16bf a, v16bf b, v8f c) {
  // v_wmma_f32_16x16x32_bf16  D = A(16x32) * B(32x16) + C
  return __builtin_amdgcn_wmma_f32_16x16x32_bf16(false, a, false, b, (short)0, c,
                                                 false, false);
}

// A-operand K index pattern for 16-bit 16x32 tiles (ISA 7.12.2):
// lanes 0-15: VGPR j -> K = {0,2,4,6,16,18,20,22}[j]; lanes 16-31: +8
__device__ __forceinline__ int a_k(int j, int hi) {
  return ((j < 4) ? (2 * j) : (16 + 2 * (j - 4))) + hi * 8;
}

// Build bf16 A operand from fp32 row-major X [rows x ldx], optional per-row scale.
__device__ __forceinline__ v16bf buildA_f32(const float* __restrict__ X, int ldx,
                                            int rowBase, int koff,
                                            const float* __restrict__ scale,
                                            int lane) {
  int m = lane & 15, hi = lane >> 4;
  int row = rowBase + m;
  float sc = scale ? scale[row] : 1.0f;
  const float* p = X + (long long)row * ldx + koff;
  v16bf a;
#pragma unroll
  for (int j = 0; j < 8; ++j) {
    int k = a_k(j, hi);
    a[2 * j]     = (__bf16)(p[k] * sc);
    a[2 * j + 1] = (__bf16)(p[k + 1] * sc);
  }
  return a;
}

// Build bf16 A operand from bf16 row-major X (pairs are byte-contiguous -> b128 loads).
__device__ __forceinline__ v16bf buildA_bf16(const __bf16* __restrict__ X,
                                             int ldx, int rowBase, int koff,
                                             int lane) {
  int m = lane & 15, hi = lane >> 4;
  const __bf16* p = X + (long long)(rowBase + m) * ldx + koff;
  V16BF a;
#pragma unroll
  for (int j = 0; j < 8; ++j)
    a.u[j] = *(const unsigned int*)(p + a_k(j, hi));   // k even -> 4B aligned
  return a.v;
}

// Load pre-swizzled B operand: Wb[((tile*KC + kc)*32 + lane)*8 + j] (32B/lane, coalesced)
__device__ __forceinline__ v16bf loadB(const unsigned int* __restrict__ Wb,
                                       int tile, int kc, int KC, int lane) {
  const unsigned int* p = Wb + ((size_t)(tile * KC + kc) * 32 + lane) * 8;
  V16BF b;
#pragma unroll
  for (int j = 0; j < 8; ++j) b.u[j] = p[j];
  return b.v;
}

// ---------------- weight pre-swizzle (bf16 B-operand layout) ----------------
__device__ __forceinline__ void prep_one(unsigned int* dst, int w, int KC,
                                         const float* src, int transposed, int ld) {
  int j = w & 7, lane = (w >> 3) & 31, rest = w >> 8;
  int kc = rest % KC, tile = rest / KC;
  int n = tile * 16 + (lane & 15);
  int k = kc * 32 + ((j < 4) ? (2 * j) : (16 + 2 * (j - 4))) + (lane >> 4) * 8;
  float lo, hi;
  if (transposed) { lo = src[n * ld + k];  hi = src[n * ld + k + 1]; }   // B = W^T, W[cols][K]
  else            { lo = src[k * ld + n];  hi = src[(k + 1) * ld + n]; } // B = W,   W[K][cols]
  dst[w] = pack_bf2(lo, hi);
}

__global__ void __launch_bounds__(256)
prep_weights(const float* __restrict__ W_ih, const float* __restrict__ W_hh,
             const float* __restrict__ W1, const float* __restrict__ W2,
             unsigned int* WbIH, unsigned int* WbHH,
             unsigned int* Wb1, unsigned int* Wb2) {
  int w = blockIdx.x * blockDim.x + threadIdx.x;
  if (w < 16384)                 prep_one(WbIH, w,          4, W_ih, 1, IN_DIM); // [256x128]^T
  else if (w < 16384 + 8192)     prep_one(WbHH, w - 16384,  2, W_hh, 1, H_DIM);  // [256x64]^T
  else if (w < 16384 + 16384)    prep_one(Wb1,  w - 24576,  2, W1,   0, NH_DIM); // [64x256]
  else if (w < 16384 + 16384 + 2048) prep_one(Wb2, w - 32768, 8, W2, 0, NC_DIM); // [256x16]
}

// ---------------- G table: G[v][:] = emb[v] @ W_ih^T + b_ih + b_hh ----------------
__global__ void __launch_bounds__(256)
gemm_table(const float* __restrict__ emb, const unsigned int* __restrict__ WbIH,
           const float* __restrict__ b_ih, const float* __restrict__ b_hh,
           float* __restrict__ G) {
  int wave = (blockIdx.x * blockDim.x + threadIdx.x) >> 5;  // exactly 2000*16 waves
  int lane = threadIdx.x & 31;
  int rowBase = (wave >> 4) * 16;
  int T = wave & 15;
  v8f acc = {};
#pragma unroll
  for (int kc = 0; kc < 4; ++kc) {
    v16bf a = buildA_f32(emb, IN_DIM, rowBase, kc * 32, nullptr, lane);
    v16bf b = loadB(WbIH, T, kc, 4, lane);
    acc = wmma_bf16(a, b, acc);
  }
  int m = lane & 15, hi = lane >> 4;
  int col = T * 16 + m;
  float bias = b_ih[col] + b_hh[col];
#pragma unroll
  for (int r = 0; r < 8; ++r)
    G[(size_t)(rowBase + r + hi * 8) * G4H + col] = acc[r] + bias;
}

// ---------------- fused LSTM step: gates = G[tok] + h @ W_hh^T; update h,c --------
__global__ void __launch_bounds__(256)
lstm_step(const float* __restrict__ G, const int* __restrict__ features,
          const unsigned int* __restrict__ WbHH,
          __bf16* __restrict__ hbf,              // in/out bf16 h (wave-private rows)
          float* __restrict__ h, float* __restrict__ c, int t) {
  int wave = (blockIdx.x * blockDim.x + threadIdx.x) >> 5;
  if (wave >= N_NODES / 16) return;              // wave-uniform guard
  int lane = threadIdx.x & 31;
  int m = lane & 15, hi = lane >> 4;
  int nodeBase = wave * 16;

  // init accumulators with gathered G rows (input contribution + biases)
  int tok[8];
#pragma unroll
  for (int r = 0; r < 8; ++r)
    tok[r] = features[(nodeBase + r + hi * 8) * SEQ_L + t];
  v8f acc[16];
#pragma unroll
  for (int T = 0; T < 16; ++T) {
#pragma unroll
    for (int r = 0; r < 8; ++r)
      acc[T][r] = G[(size_t)tok[r] * G4H + T * 16 + m];
  }
  // recurrent GEMM: 16 col-tiles x K=64 (2 chunks) = 32 WMMAs
#pragma unroll
  for (int kc = 0; kc < 2; ++kc) {
    v16bf a = buildA_bf16(hbf, H_DIM, nodeBase, kc * 32, lane);
#pragma unroll
    for (int T = 0; T < 16; ++T) {
      v16bf b = loadB(WbHH, T, kc, 2, lane);
      acc[T] = wmma_bf16(a, b, acc[T]);
    }
  }
  // elementwise: i,f,g,o are tiles T, T+4, T+8, T+12 at identical lane/vgpr slots
#pragma unroll
  for (int T = 0; T < 4; ++T) {
#pragma unroll
    for (int r = 0; r < 8; ++r) {
      int node = nodeBase + r + hi * 8;
      int u = T * 16 + m;
      float ig = acc[T][r], fg = acc[T + 4][r], gg = acc[T + 8][r], og = acc[T + 12][r];
      size_t idx = (size_t)node * H_DIM + u;
      float cN = sigf(fg) * c[idx] + sigf(ig) * tanhf(gg);
      float hN = sigf(og) * tanhf(cN);
      c[idx] = cN;
      h[idx] = hN;
      hbf[idx] = (__bf16)hN;
    }
  }
}

// ---------------- graph: degrees + normalization ----------------
__global__ void __launch_bounds__(256)
degrees_k(const int* __restrict__ src, const int* __restrict__ dst,
          float* degO, float* degI) {
  int e = blockIdx.x * blockDim.x + threadIdx.x;
  if (e < N_EDGES) {
    atomicAdd(&degO[src[e]], 1.0f);
    atomicAdd(&degI[dst[e]], 1.0f);
  }
}
__global__ void __launch_bounds__(256)
norm_deg(float* degO, float* degI) {
  int i = blockIdx.x * blockDim.x + threadIdx.x;
  if (i < N_NODES) {
    float d = degO[i]; degO[i] = (d > 0.0f) ? rsqrtf(d) : 1.0f;
    d = degI[i];       degI[i] = (d > 0.0f) ? rsqrtf(d) : 1.0f;
  }
}

// scatter-add: agg[dst] += X[src] * nsrc[src]; 4 feats per thread
template <int D>
__global__ void __launch_bounds__(256)
edge_agg(const int* __restrict__ src, const int* __restrict__ dst,
         const float* __restrict__ nsrc, const float* __restrict__ X,
         float* __restrict__ agg) {
  const int per = D / 4;
  long long tid = (long long)blockIdx.x * blockDim.x + threadIdx.x;
  long long e = tid / per;
  if (e >= N_EDGES) return;
  int p = (int)(tid % per) * 4;
  int s = src[e], d = dst[e];
  float ns = nsrc[s];
  float4 v = *(const float4*)(X + (size_t)s * D + p);
  float* a = agg + (size_t)d * D + p;
  atomicAdd(a + 0, v.x * ns);
  atomicAdd(a + 1, v.y * ns);
  atomicAdd(a + 2, v.z * ns);
  atomicAdd(a + 3, v.w * ns);
}

// out[rows x COLT*16] = act((A * rowScale) @ B + bias); KC = K/32 chunks
template <int KC, int COLT>
__global__ void __launch_bounds__(256)
gemm_conv(const float* __restrict__ A, const float* __restrict__ rowScale,
          const unsigned int* __restrict__ Wb, const float* __restrict__ bias,
          float* __restrict__ out, int relu) {
  int wave = (blockIdx.x * blockDim.x + threadIdx.x) >> 5;
  if (wave >= N_NODES / 16) return;              // wave-uniform guard
  int lane = threadIdx.x & 31;
  int m = lane & 15, hi = lane >> 4;
  int rowBase = wave * 16;
  const int K = KC * 32, ldOut = COLT * 16;

  v16bf a[KC];
#pragma unroll
  for (int kc = 0; kc < KC; ++kc)
    a[kc] = buildA_f32(A, K, rowBase, kc * 32, rowScale, lane);

#pragma unroll
  for (int T = 0; T < COLT; ++T) {
    v8f acc = {};
#pragma unroll
    for (int kc = 0; kc < KC; ++kc) {
      v16bf b = loadB(Wb, T, kc, KC, lane);
      acc = wmma_bf16(a[kc], b, acc);
    }
    int col = T * 16 + m;
    float bs = bias[col];
#pragma unroll
    for (int r = 0; r < 8; ++r) {
      float v = acc[r] + bs;
      if (relu) v = fmaxf(v, 0.0f);
      out[(size_t)(rowBase + r + hi * 8) * ldOut + col] = v;
    }
  }
}

// -------------------------------- host --------------------------------
extern "C" void kernel_launch(void* const* d_in, const int* in_sizes, int n_in,
                              void* d_out, int out_size, void* d_ws, size_t ws_size,
                              hipStream_t stream) {
  (void)in_sizes; (void)n_in; (void)out_size; (void)ws_size;
  const int*   features = (const int*)d_in[0];
  const int*   src      = (const int*)d_in[1];
  const int*   dst      = (const int*)d_in[2];
  const float* emb      = (const float*)d_in[3];
  const float* W_ih     = (const float*)d_in[4];
  const float* W_hh     = (const float*)d_in[5];
  const float* b_ih     = (const float*)d_in[6];
  const float* b_hh     = (const float*)d_in[7];
  const float* W1       = (const float*)d_in[8];
  const float* b1       = (const float*)d_in[9];
  const float* W2       = (const float*)d_in[10];
  const float* b2       = (const float*)d_in[11];
  float* out = (float*)d_out;

  char* ws = (char*)d_ws;
  size_t off = 0;
  auto take = [&](size_t bytes) -> char* {
    char* p = ws + off;
    off += (bytes + 255) & ~(size_t)255;
    return p;
  };
  float*        G    = (float*)take((size_t)VOCAB * G4H * 4);        // 32.8 MB
  float*        h    = (float*)take((size_t)N_NODES * H_DIM * 4);
  float*        c    = (float*)take((size_t)N_NODES * H_DIM * 4);
  __bf16*       hbf  = (__bf16*)take((size_t)N_NODES * H_DIM * 2);
  unsigned int* WbIH = (unsigned int*)take(16384 * 4);
  unsigned int* WbHH = (unsigned int*)take(8192 * 4);
  unsigned int* Wb1  = (unsigned int*)take(8192 * 4);
  unsigned int* Wb2  = (unsigned int*)take(2048 * 4);
  float*        nsrc = (float*)take((size_t)N_NODES * 4);
  float*        ndst = (float*)take((size_t)N_NODES * 4);
  float*        agg1 = (float*)take((size_t)N_NODES * H_DIM * 4);
  float*        out1 = (float*)take((size_t)N_NODES * NH_DIM * 4);
  float*        agg2 = (float*)take((size_t)N_NODES * NH_DIM * 4);

  hipMemsetAsync(c,    0, (size_t)N_NODES * H_DIM * 4, stream);
  hipMemsetAsync(hbf,  0, (size_t)N_NODES * H_DIM * 2, stream);
  hipMemsetAsync(nsrc, 0, (size_t)N_NODES * 4, stream);
  hipMemsetAsync(ndst, 0, (size_t)N_NODES * 4, stream);
  hipMemsetAsync(agg1, 0, (size_t)N_NODES * H_DIM * 4, stream);
  hipMemsetAsync(agg2, 0, (size_t)N_NODES * NH_DIM * 4, stream);

  prep_weights<<<(34816 + 255) / 256, 256, 0, stream>>>(W_ih, W_hh, W1, W2,
                                                        WbIH, WbHH, Wb1, Wb2);
  // G = emb @ W_ih^T + b: 2000*16 waves, 8 waves/block -> exactly 4000 blocks
  gemm_table<<<4000, 256, 0, stream>>>(emb, WbIH, b_ih, b_hh, G);

  // 1250 row-panel waves per step; 157 blocks of 8 waves (guarded)
  for (int t = 0; t < SEQ_L; ++t)
    lstm_step<<<157, 256, 0, stream>>>(G, features, WbHH, hbf, h, c, t);

  degrees_k<<<(N_EDGES + 255) / 256, 256, 0, stream>>>(src, dst, nsrc, ndst);
  norm_deg<<<(N_NODES + 255) / 256, 256, 0, stream>>>(nsrc, ndst);

  // GraphConv 1: agg1 = scatter(h*nsrc); out1 = relu((agg1*ndst) @ W1 + b1)
  {
    long long th = (long long)N_EDGES * (H_DIM / 4);
    edge_agg<H_DIM><<<(unsigned)((th + 255) / 256), 256, 0, stream>>>(src, dst, nsrc, h, agg1);
  }
  gemm_conv<2, 16><<<157, 256, 0, stream>>>(agg1, ndst, Wb1, b1, out1, 1);

  // GraphConv 2: agg2 = scatter(out1*nsrc); out = (agg2*ndst) @ W2 + b2
  {
    long long th = (long long)N_EDGES * (NH_DIM / 4);
    edge_agg<NH_DIM><<<(unsigned)((th + 255) / 256), 256, 0, stream>>>(src, dst, nsrc, out1, agg2);
  }
  gemm_conv<8, 1><<<157, 256, 0, stream>>>(agg2, ndst, Wb2, b2, out, 0);
}
